// GGANN_31387620999818
// MI455X (gfx1250) — compile-verified, matrix operand order checked by reading, not run on hardware
//
#include <hip/hip_runtime.h>
#include <math.h>

typedef _Float16 half_t;
typedef __attribute__((ext_vector_type(16))) _Float16 v16h;
typedef __attribute__((ext_vector_type(8)))  _Float16 v8h;
typedef __attribute__((ext_vector_type(8)))  float    v8f;

#define SD    64
#define MD    64
#define HM    128
#define HR    128
#define NSTEPS 10
#define NVARS 2048
#define KPAD  160    // padded layer-1 input dim (133/132 -> 160, 5 K-chunks of 32)
#define ETILE 32     // edges per block in edge-MLP kernel

// ---- WMMA fragment helpers (wave32, 16x16x32 f16) -------------------------
// A-matrix 16x32 f16 (ISA 7.12.2): lane L holds row m=L&15; halves 0..7 are
// k = (L>>4)*8 + 0..7, halves 8..15 are k = (L>>4)*8 + 16..23. Both runs are
// contiguous in a k-major row -> two 16B LDS loads per fragment.
__device__ __forceinline__ v16h fragA_lds(const half_t* X, int stride, int kbase) {
    int lane = threadIdx.x & 31;
    const half_t* p = X + (lane & 15) * stride + kbase + ((lane >> 4) << 3);
    v8h lo = *(const v8h*)p;
    v8h hi = *(const v8h*)(p + 16);
    return __builtin_shufflevector(lo, hi, 0, 1, 2, 3, 4, 5, 6, 7,
                                           8, 9, 10, 11, 12, 13, 14, 15);
}

// B fragments pre-swizzled in global memory: each lane's 16 halves contiguous.
__device__ __forceinline__ v16h fragB_sw(const half_t* __restrict__ W, int NT, int kc, int nt) {
    int lane = threadIdx.x & 31;
    return *(const v16h*)(W + ((((size_t)kc * NT + nt) * 32 + lane) << 4));
}

__device__ __forceinline__ v8f wmma_f16(v16h a, v16h b, v8f c) {
    return __builtin_amdgcn_wmma_f32_16x16x32_f16(false, a, false, b, (short)0, c, false, false);
}

// ---- utility kernels ------------------------------------------------------
__global__ void zero_kernel(float* __restrict__ p, int n) {
    int i = blockIdx.x * blockDim.x + threadIdx.x;
    if (i < n) p[i] = 0.f;
}

// W [OUT,IN] fp32 -> fragment-swizzled f16: dst[((kc*NT+nt)*32+lane)*16+h]
__global__ void convert_swz_kernel(half_t* __restrict__ dst, const float* __restrict__ src,
                                   int OUT, int IN, int KP) {
    int i = blockIdx.x * blockDim.x + threadIdx.x;
    if (i >= KP * OUT) return;
    int h = i & 15, lane = (i >> 4) & 31, rest = i >> 9;
    int NT = OUT >> 4;
    int nt = rest % NT, kc = rest / NT;
    int k = kc * 32 + ((lane >> 4) << 3) + ((h >> 3) << 4) + (h & 7);
    int n = nt * 16 + (lane & 15);
    dst[i] = (half_t)((k < IN) ? src[n * IN + k] : 0.f);
}

// ---- attention logits: one wave (32 lanes) per edge -----------------------
__global__ void __launch_bounds__(256)
logits_kernel(const float* __restrict__ hA, const float* __restrict__ hB,
              const int* __restrict__ rowIdx, const int* __restrict__ colIdx,
              const float* __restrict__ attnW, const float* __restrict__ attnb,
              float* __restrict__ elog, int E) {
    int wave = threadIdx.x >> 5, lane = threadIdx.x & 31;
    int ge = blockIdx.x * 8 + wave;
    if (ge >= E) return;
    const float* hi = hA + (size_t)rowIdx[ge] * SD;
    const float* hj = hB + (size_t)colIdx[ge] * SD;
    float s = hi[lane] * attnW[lane] + hi[lane + 32] * attnW[lane + 32]
            + hj[lane] * attnW[lane + 64] + hj[lane + 32] * attnW[lane + 96];
#pragma unroll
    for (int off = 16; off >= 1; off >>= 1) s += __shfl_xor(s, off, 32);
    if (lane == 0) {
        s += attnb[0];
        elog[ge] = (s > 0.f) ? s : 0.01f * s;   // leaky_relu
    }
}

// ---- global softmax stats: single workgroup, whole edge set ---------------
__global__ void __launch_bounds__(1024)
softmax_reduce_kernel(const float* __restrict__ elog, int E, float* __restrict__ red) {
    __shared__ float sm[1024];
    int tid = threadIdx.x;
    float m = -3.0e38f;
    for (int i = tid; i < E; i += 1024) m = fmaxf(m, elog[i]);
    sm[tid] = m; __syncthreads();
    for (int s = 512; s >= 1; s >>= 1) {
        if (tid < s) sm[tid] = fmaxf(sm[tid], sm[tid + s]);
        __syncthreads();
    }
    float gmax = sm[0]; __syncthreads();
    float acc = 0.f;
    for (int i = tid; i < E; i += 1024) acc += __expf(elog[i] - gmax);
    sm[tid] = acc; __syncthreads();
    for (int s = 512; s >= 1; s >>= 1) {
        if (tid < s) sm[tid] += sm[tid + s];
        __syncthreads();
    }
    if (tid == 0) { red[0] = gmax; red[1] = sm[0]; }
}

// ---- fused edge MLP (3 layers, WMMA) + softmax scale + segment scatter ----
// 32 edges per block: each B fragment feeds two 16-row A tiles.
__global__ void __launch_bounds__(256)
edge_mlp_kernel(const float* __restrict__ hA, const float* __restrict__ hB,
                const float* __restrict__ feat, int featn,
                const int* __restrict__ rowIdx, const int* __restrict__ colIdx,
                const half_t* __restrict__ W1, const float* __restrict__ b1,
                const half_t* __restrict__ W2, const float* __restrict__ b2,
                const half_t* __restrict__ W3, const float* __restrict__ b3,
                const float* __restrict__ elog, const float* __restrict__ red,
                float* __restrict__ seg, int E) {
    __shared__ __align__(32) half_t X[ETILE * KPAD];
    __shared__ __align__(32) half_t H1[ETILE * HM];
    __shared__ __align__(32) half_t H2[ETILE * HM];
    __shared__ float coefs[ETILE];
    __shared__ int   cols[ETILE];
    __shared__ int   rows[ETILE];
    int tile = blockIdx.x, tid = threadIdx.x;

    if (tid < ETILE) {
        int ge = tile * ETILE + tid;
        float c = 0.f; int col = 0, row = 0;
        if (ge < E) { c = __expf(elog[ge] - red[0]) / red[1]; col = colIdx[ge]; row = rowIdx[ge]; }
        coefs[tid] = c; cols[tid] = col; rows[tid] = row;
    }
    __syncthreads();

    // X[:, 0:128) = [hi | hj] via float4 gathers (4 iters/thread)
    for (int i = tid; i < ETILE * 32; i += 256) {
        int m = i >> 5, q = i & 31;
        int ge = tile * ETILE + m;
        float4 v = {0.f, 0.f, 0.f, 0.f};
        if (ge < E) {
            if (q < 16) v = ((const float4*)(hA + (size_t)rows[m] * SD))[q];
            else        v = ((const float4*)(hB + (size_t)cols[m] * SD))[q - 16];
        }
        half_t* d = &X[m * KPAD + (q << 2)];
        d[0] = (half_t)v.x; d[1] = (half_t)v.y; d[2] = (half_t)v.z; d[3] = (half_t)v.w;
    }
    // X[:, 128:160) = [feat | 0-pad]
    for (int i = tid; i < ETILE * 32; i += 256) {
        int m = i >> 5, kk = i & 31;
        int ge = tile * ETILE + m;
        float v = 0.f;
        if (ge < E && kk < featn) v = feat[(size_t)ge * featn + kk];
        X[m * KPAD + 128 + kk] = (half_t)v;
    }
    __syncthreads();

    int wave = tid >> 5, lane = tid & 31;
    // layer 1: [32 x KPAD] @ [KPAD x 128]
    {
        v8f a0 = {}, a1 = {};
#pragma unroll
        for (int kc = 0; kc < KPAD / 32; ++kc) {
            v16h b = fragB_sw(W1, HM / 16, kc, wave);
            a0 = wmma_f16(fragA_lds(X, KPAD, kc * 32), b, a0);
            a1 = wmma_f16(fragA_lds(X + 16 * KPAD, KPAD, kc * 32), b, a1);
        }
        int n = wave * 16 + (lane & 15);
        float bias = b1[n];
#pragma unroll
        for (int v = 0; v < 8; ++v) {
            int m = v + ((lane >> 4) << 3);
            float r0 = a0[v] + bias; if (r0 < 0.f) r0 = 0.f;
            float r1 = a1[v] + bias; if (r1 < 0.f) r1 = 0.f;
            H1[m * HM + n] = (half_t)r0;
            H1[(m + 16) * HM + n] = (half_t)r1;
        }
    }
    __syncthreads();
    // layer 2: [32 x 128] @ [128 x 128]
    {
        v8f a0 = {}, a1 = {};
#pragma unroll
        for (int kc = 0; kc < HM / 32; ++kc) {
            v16h b = fragB_sw(W2, HM / 16, kc, wave);
            a0 = wmma_f16(fragA_lds(H1, HM, kc * 32), b, a0);
            a1 = wmma_f16(fragA_lds(H1 + 16 * HM, HM, kc * 32), b, a1);
        }
        int n = wave * 16 + (lane & 15);
        float bias = b2[n];
#pragma unroll
        for (int v = 0; v < 8; ++v) {
            int m = v + ((lane >> 4) << 3);
            float r0 = a0[v] + bias; if (r0 < 0.f) r0 = 0.f;
            float r1 = a1[v] + bias; if (r1 < 0.f) r1 = 0.f;
            H2[m * HM + n] = (half_t)r0;
            H2[(m + 16) * HM + n] = (half_t)r1;
        }
    }
    __syncthreads();
    // layer 3: [32 x 128] @ [128 x 64] on waves 0..3; scale + scatter
    if (wave < 4) {
        v8f a0 = {}, a1 = {};
#pragma unroll
        for (int kc = 0; kc < HM / 32; ++kc) {
            v16h b = fragB_sw(W3, MD / 16, kc, wave);
            a0 = wmma_f16(fragA_lds(H2, HM, kc * 32), b, a0);
            a1 = wmma_f16(fragA_lds(H2 + 16 * HM, HM, kc * 32), b, a1);
        }
        int n = wave * 16 + (lane & 15);
        float bias = b3[n];
#pragma unroll
        for (int v = 0; v < 8; ++v) {
            int m = v + ((lane >> 4) << 3);
            int ge0 = tile * ETILE + m;
            int ge1 = ge0 + 16;
            if (ge0 < E) atomicAdd(&seg[(size_t)cols[m] * MD + n],      (a0[v] + bias) * coefs[m]);
            if (ge1 < E) atomicAdd(&seg[(size_t)cols[m + 16] * MD + n], (a1[v] + bias) * coefs[m + 16]);
        }
    }
}

// ---- GRU update (WMMA for gi/gh projections, VALU for gates) --------------
__global__ void __launch_bounds__(256)
gru_kernel(const float* __restrict__ seg, float* __restrict__ h,
           const half_t* __restrict__ Wi, const half_t* __restrict__ Wh,
           const float* __restrict__ bi, const float* __restrict__ bh, int M) {
    __shared__ __align__(32) half_t Xi[16 * SD];
    __shared__ __align__(32) half_t Xh[16 * SD];
    __shared__ float Gi[16 * 3 * SD];
    __shared__ float Gh[16 * 3 * SD];
    int tile = blockIdx.x, tid = threadIdx.x;
    for (int i = tid; i < 16 * (SD / 4); i += 256) {   // float4 staging
        int m = i >> 4, q = i & 15;
        size_t g = (size_t)(tile * 16 + m) * SD + (q << 2);
        float4 vi = *(const float4*)(seg + g);
        float4 vh = *(const float4*)(h + g);
        half_t* di = &Xi[m * SD + (q << 2)];
        half_t* dh = &Xh[m * SD + (q << 2)];
        di[0] = (half_t)vi.x; di[1] = (half_t)vi.y; di[2] = (half_t)vi.z; di[3] = (half_t)vi.w;
        dh[0] = (half_t)vh.x; dh[1] = (half_t)vh.y; dh[2] = (half_t)vh.z; dh[3] = (half_t)vh.w;
    }
    __syncthreads();
    int wave = tid >> 5, lane = tid & 31;
    for (int nt = wave; nt < 3 * SD / 16; nt += 8) {   // 12 N-tiles
        v8f ai = {}, ah = {};
#pragma unroll
        for (int kc = 0; kc < SD / 32; ++kc) {
            ai = wmma_f16(fragA_lds(Xi, SD, kc * 32), fragB_sw(Wi, 3 * SD / 16, kc, nt), ai);
            ah = wmma_f16(fragA_lds(Xh, SD, kc * 32), fragB_sw(Wh, 3 * SD / 16, kc, nt), ah);
        }
        int n = nt * 16 + (lane & 15);
        float bii = bi[n], bhh = bh[n];
#pragma unroll
        for (int v = 0; v < 8; ++v) {
            int m = v + ((lane >> 4) << 3);
            Gi[m * 3 * SD + n] = ai[v] + bii;
            Gh[m * 3 * SD + n] = ah[v] + bhh;
        }
    }
    __syncthreads();
    for (int i = tid; i < 16 * SD; i += 256) {
        int m = i / SD, d = i - m * SD;
        size_t g = (size_t)(tile * 16 + m) * SD + d;
        float r = 1.f / (1.f + __expf(-(Gi[m * 192 + d]      + Gh[m * 192 + d])));
        float z = 1.f / (1.f + __expf(-(Gi[m * 192 + 64 + d] + Gh[m * 192 + 64 + d])));
        float nn = tanhf(Gi[m * 192 + 128 + d] + r * Gh[m * 192 + 128 + d]);
        float hold = h[g];
        h[g] = (1.f - z) * nn + z * hold;
    }
}

// ---- readout --------------------------------------------------------------
__global__ void readout_scatter_kernel(const float* __restrict__ h, const int* __restrict__ col,
                                       float* __restrict__ node, int M) {
    int i = blockIdx.x * blockDim.x + threadIdx.x;
    if (i >= M * SD) return;
    int m = i / SD, d = i - m * SD;
    atomicAdd(&node[(size_t)col[m] * SD + d], h[i]);
}

__global__ void __launch_bounds__(256)
readout_mlp_kernel(const float* __restrict__ node,
                   const half_t* __restrict__ W1, const float* __restrict__ b1,
                   const half_t* __restrict__ W2, const float* __restrict__ b2,
                   const float* __restrict__ W3, const float* __restrict__ b3,
                   float* __restrict__ out) {
    __shared__ __align__(32) half_t X[16 * SD];
    __shared__ __align__(32) half_t H1[16 * HR];
    __shared__ __align__(32) half_t H2[16 * HR];
    int tile = blockIdx.x, tid = threadIdx.x;
    for (int i = tid; i < 16 * (SD / 4); i += 256) {
        int m = i >> 4, q = i & 15;
        float4 v = *(const float4*)(node + (size_t)(tile * 16 + m) * SD + (q << 2));
        half_t* d = &X[m * SD + (q << 2)];
        d[0] = (half_t)v.x; d[1] = (half_t)v.y; d[2] = (half_t)v.z; d[3] = (half_t)v.w;
    }
    __syncthreads();
    int wave = tid >> 5, lane = tid & 31;
    {
        v8f acc = {};
#pragma unroll
        for (int kc = 0; kc < SD / 32; ++kc)
            acc = wmma_f16(fragA_lds(X, SD, kc * 32), fragB_sw(W1, HR / 16, kc, wave), acc);
        int n = wave * 16 + (lane & 15);
        float bias = b1[n];
#pragma unroll
        for (int v = 0; v < 8; ++v) {
            int m = v + ((lane >> 4) << 3);
            float r = acc[v] + bias; if (r < 0.f) r = 0.f;
            H1[m * HR + n] = (half_t)r;
        }
    }
    __syncthreads();
    {
        v8f acc = {};
#pragma unroll
        for (int kc = 0; kc < HR / 32; ++kc)
            acc = wmma_f16(fragA_lds(H1, HR, kc * 32), fragB_sw(W2, HR / 16, kc, wave), acc);
        int n = wave * 16 + (lane & 15);
        float bias = b2[n];
#pragma unroll
        for (int v = 0; v < 8; ++v) {
            int m = v + ((lane >> 4) << 3);
            float r = acc[v] + bias; if (r < 0.f) r = 0.f;
            H2[m * HR + n] = (half_t)r;
        }
    }
    __syncthreads();
    if (tid < 16) {   // final 128 -> 2 + softmax(axis=1)
        float l0 = b3[0], l1 = b3[1];
        for (int k = 0; k < HR; ++k) {
            float hv = (float)H2[tid * HR + k];
            l0 += hv * W3[k];
            l1 += hv * W3[HR + k];
        }
        float mx = fmaxf(l0, l1);
        float e0 = __expf(l0 - mx), e1 = __expf(l1 - mx);
        float inv = 1.f / (e0 + e1);
        out[(size_t)(tile * 16 + tid) * 2 + 0] = e0 * inv;
        out[(size_t)(tile * 16 + tid) * 2 + 1] = e1 * inv;
    }
}

// ---- host orchestration ---------------------------------------------------
extern "C" void kernel_launch(void* const* d_in, const int* in_sizes, int n_in,
                              void* d_out, int out_size, void* d_ws, size_t ws_size,
                              hipStream_t stream) {
    const float* attn_W = (const float*)d_in[0];
    const float* attn_b = (const float*)d_in[1];
    const float* v2f_W1 = (const float*)d_in[2];  const float* v2f_b1 = (const float*)d_in[3];
    const float* v2f_W2 = (const float*)d_in[4];  const float* v2f_b2 = (const float*)d_in[5];
    const float* v2f_W3 = (const float*)d_in[6];  const float* v2f_b3 = (const float*)d_in[7];
    const float* f2v_W1 = (const float*)d_in[8];  const float* f2v_b1 = (const float*)d_in[9];
    const float* f2v_W2 = (const float*)d_in[10]; const float* f2v_b2 = (const float*)d_in[11];
    const float* f2v_W3 = (const float*)d_in[12]; const float* f2v_b3 = (const float*)d_in[13];
    const float* ro_W1  = (const float*)d_in[14]; const float* ro_b1  = (const float*)d_in[15];
    const float* ro_W2  = (const float*)d_in[16]; const float* ro_b2  = (const float*)d_in[17];
    const float* ro_W3  = (const float*)d_in[18]; const float* ro_b3  = (const float*)d_in[19];
    const float* gv_Wi  = (const float*)d_in[20]; const float* gv_Wh  = (const float*)d_in[21];
    const float* gv_bi  = (const float*)d_in[22]; const float* gv_bh  = (const float*)d_in[23];
    const float* gf_Wi  = (const float*)d_in[24]; const float* gf_Wh  = (const float*)d_in[25];
    const float* gf_bi  = (const float*)d_in[26]; const float* gf_bh  = (const float*)d_in[27];
    const float* feat5  = (const float*)d_in[28];
    const float* feat4  = (const float*)d_in[29];
    const int*   f2v_row = (const int*)d_in[30];
    const int*   f2v_col = (const int*)d_in[31];
    const int*   v2f_row = (const int*)d_in[32];
    const int*   v2f_col = (const int*)d_in[33];
    const int*   ro_col  = (const int*)d_in[34];

    const int E = in_sizes[30];           // f2v edges
    const int M = in_sizes[32];           // v2f edges == message nodes
    const int EMAX = (E > M) ? E : M;

    char* w = (char*)d_ws;
    auto alloc = [&](size_t bytes) -> void* {
        void* p = (void*)w; w += (bytes + 255) & ~(size_t)255; return p;
    };
    float*  h_f2v = (float*)alloc((size_t)M * SD * 4);
    float*  h_v2f = (float*)alloc((size_t)M * SD * 4);
    float*  seg   = (float*)alloc((size_t)M * SD * 4);
    float*  elog  = (float*)alloc((size_t)EMAX * 4);
    float*  red   = (float*)alloc(2 * 4);
    float*  node  = (float*)alloc((size_t)NVARS * SD * 4);
    half_t* w1v  = (half_t*)alloc((size_t)KPAD * HM * 2);
    half_t* w2v  = (half_t*)alloc((size_t)HM * HM * 2);
    half_t* w3v  = (half_t*)alloc((size_t)HM * MD * 2);
    half_t* w1f  = (half_t*)alloc((size_t)KPAD * HM * 2);
    half_t* w2f  = (half_t*)alloc((size_t)HM * HM * 2);
    half_t* w3f  = (half_t*)alloc((size_t)HM * MD * 2);
    half_t* wro1 = (half_t*)alloc((size_t)SD * HR * 2);
    half_t* wro2 = (half_t*)alloc((size_t)HR * HR * 2);
    half_t* wgvi = (half_t*)alloc((size_t)SD * 3 * SD * 2);
    half_t* wgvh = (half_t*)alloc((size_t)SD * 3 * SD * 2);
    half_t* wgfi = (half_t*)alloc((size_t)SD * 3 * SD * 2);
    half_t* wgfh = (half_t*)alloc((size_t)SD * 3 * SD * 2);

    auto cvt = [&](half_t* dst, const float* src, int OUT, int IN, int KP) {
        int n = KP * OUT;
        convert_swz_kernel<<<(n + 255) / 256, 256, 0, stream>>>(dst, src, OUT, IN, KP);
    };
    cvt(w1v, v2f_W1, HM, 2 * SD + 5, KPAD);
    cvt(w2v, v2f_W2, HM, HM, HM);
    cvt(w3v, v2f_W3, MD, HM, HM);
    cvt(w1f, f2v_W1, HM, 2 * SD + 4, KPAD);
    cvt(w2f, f2v_W2, HM, HM, HM);
    cvt(w3f, f2v_W3, MD, HM, HM);
    cvt(wro1, ro_W1, HR, SD, SD);
    cvt(wro2, ro_W2, HR, HR, HR);
    cvt(wgvi, gv_Wi, 3 * SD, SD, SD);
    cvt(wgvh, gv_Wh, 3 * SD, SD, SD);
    cvt(wgfi, gf_Wi, 3 * SD, SD, SD);
    cvt(wgfh, gf_Wh, 3 * SD, SD, SD);

    const int hse = M * SD;
    zero_kernel<<<(hse + 255) / 256, 256, 0, stream>>>(h_f2v, hse);
    zero_kernel<<<(hse + 255) / 256, 256, 0, stream>>>(h_v2f, hse);

    for (int step = 0; step < NSTEPS; ++step) {
        // ---- direction A: update v2f_h (hi=f2v_h[row], hj=v2f_h[col], feat5)
        logits_kernel<<<(E + 7) / 8, 256, 0, stream>>>(h_f2v, h_v2f, f2v_row, f2v_col,
                                                       attn_W, attn_b, elog, E);
        softmax_reduce_kernel<<<1, 1024, 0, stream>>>(elog, E, red);
        zero_kernel<<<(hse + 255) / 256, 256, 0, stream>>>(seg, hse);
        edge_mlp_kernel<<<(E + ETILE - 1) / ETILE, 256, 0, stream>>>(
            h_f2v, h_v2f, feat5, 5, f2v_row, f2v_col,
            w1v, v2f_b1, w2v, v2f_b2, w3v, v2f_b3, elog, red, seg, E);
        gru_kernel<<<M / 16, 256, 0, stream>>>(seg, h_v2f, wgvi, wgvh, gv_bi, gv_bh, M);

        // ---- direction B: update f2v_h (hi=v2f_h[row], hj=f2v_h[col], feat4)
        logits_kernel<<<(M + 7) / 8, 256, 0, stream>>>(h_v2f, h_f2v, v2f_row, v2f_col,
                                                       attn_W, attn_b, elog, M);
        softmax_reduce_kernel<<<1, 1024, 0, stream>>>(elog, M, red);
        zero_kernel<<<(hse + 255) / 256, 256, 0, stream>>>(seg, hse);
        edge_mlp_kernel<<<(M + ETILE - 1) / ETILE, 256, 0, stream>>>(
            h_v2f, h_f2v, feat4, 4, v2f_row, v2f_col,
            w1f, f2v_b1, w2f, f2v_b2, w3f, f2v_b3, elog, red, seg, M);
        gru_kernel<<<M / 16, 256, 0, stream>>>(seg, h_f2v, wgfi, wgfh, gf_bi, gf_bh, M);
    }

    // ---- readout
    const int nn = NVARS * SD;
    zero_kernel<<<(nn + 255) / 256, 256, 0, stream>>>(node, nn);
    readout_scatter_kernel<<<(hse + 255) / 256, 256, 0, stream>>>(h_f2v, ro_col, node, M);
    readout_mlp_kernel<<<NVARS / 16, 256, 0, stream>>>(node, wro1, ro_b1, wro2, ro_b2,
                                                       ro_W3, ro_b3, (float*)d_out);
}
